// MMSingleStreamBlock_34256659153521
// MI455X (gfx1250) — matrix operand order repaired and checked by reading, not compiled, not run on hardware
//
#include <hip/hip_runtime.h>

typedef __attribute__((ext_vector_type(16))) __bf16 v16bf;
typedef __attribute__((ext_vector_type(8)))  __bf16 v8bf;
typedef __attribute__((ext_vector_type(8)))  float  v8f;

#define HID    3072
#define HEADS  24
#define HD     128
#define SEQ    2304
#define IMGL   2048
#define N1     21504   // 3*HID + MLP
#define QKV_N  9216    // 3*HID
#define K2     15360   // HID + MLP

// ---------- helpers ----------
__device__ inline __bf16 f2bf(float f) {
  union { float f; unsigned u; } a; a.f = f;
  unsigned r = a.u + 0x7FFFu + ((a.u >> 16) & 1u);   // round-to-nearest-even
  unsigned short h = (unsigned short)(r >> 16);
  return __builtin_bit_cast(__bf16, h);
}

__device__ inline v16bf ld_ab(const __bf16* lo, const __bf16* hi) {
  union { v16bf v; v8bf h[2]; } u;
  u.h[0] = *(const v8bf*)lo;
  u.h[1] = *(const v8bf*)hi;
  return u.v;
}

__device__ inline v8f wmma_bf16(v16bf a, v16bf b, v8f c) {
  return __builtin_amdgcn_wmma_f32_16x16x32_bf16(false, a, false, b, (short)0, c, false, false);
}

__device__ inline float gelu_tanh(float x) {
  return 0.5f * x * (1.0f + tanhf(0.7978845608028654f * (x + 0.044715f * x * x * x)));
}

// low 32 bits of a generic pointer to LDS == wave-relative LDS byte address
__device__ inline unsigned ldsaddr(const void* p) {
  return (unsigned)(size_t)p;
}

// CDNA5 async global->LDS DMA, 16B per lane, tracked by ASYNCcnt
__device__ inline void async_cp16(unsigned lds, const __bf16* g) {
  asm volatile("global_load_async_to_lds_b128 %0, %1, off"
               :: "v"(lds), "v"((unsigned long long)(size_t)g)
               : "memory");
}

__device__ inline void wait_async0() {
  asm volatile("s_wait_asynccnt 0" ::: "memory");
}

// ---------- 1) mod = silu(vec) @ w_mod + b_mod (split-K GEMV) ----------
__global__ void zero_mod_kernel(float* mod) {
  mod[blockIdx.x * 256 + threadIdx.x] = 0.0f;
}

__global__ void mod_gemv_kernel(const float* __restrict__ vec,
                                const float* __restrict__ w_mod,
                                const float* __restrict__ b_mod,
                                float* __restrict__ mod) {
  int j = blockIdx.x * 128 + threadIdx.x;          // 72 blocks * 128 = 9216
  int ks = blockIdx.y * 192;                       // 16 slices of 192
  float acc = 0.0f;
  for (int k = ks; k < ks + 192; ++k) {
    float v = vec[k];
    float s = v / (1.0f + __expf(-v));             // silu
    acc += s * w_mod[(size_t)k * QKV_N + j];
  }
  if (blockIdx.y == 0) acc += b_mod[j];
  atomicAdd(&mod[j], acc);
}

// ---------- 2) transpose + convert f32 (KxN) -> bf16 (NxK) ----------
__global__ void transcvt_kernel(const float* __restrict__ in, __bf16* __restrict__ out,
                                int K, int N) {
  __shared__ float t[32][33];
  int n0 = blockIdx.x * 32, k0 = blockIdx.y * 32;
  int tx = threadIdx.x, ty = threadIdx.y;          // 32 x 8
#pragma unroll
  for (int j = 0; j < 4; ++j)
    t[ty + 8 * j][tx] = in[(size_t)(k0 + ty + 8 * j) * N + n0 + tx];
  __syncthreads();
#pragma unroll
  for (int j = 0; j < 4; ++j)
    out[(size_t)(n0 + ty + 8 * j) * K + k0 + tx] = f2bf(t[tx][ty + 8 * j]);
}

// ---------- 3) layernorm + modulation -> bf16 ----------
__global__ void ln_mod_kernel(const float* __restrict__ x, const float* __restrict__ mod,
                              __bf16* __restrict__ xm) {
  int row = blockIdx.x;
  const float* xr = x + (size_t)row * HID;
  int tid = threadIdx.x, wid = tid >> 5, lane = tid & 31;
  __shared__ float red[2][8];
  float s = 0.f, ss = 0.f;
  for (int i = tid; i < HID; i += 256) { float v = xr[i]; s += v; ss += v * v; }
  for (int off = 16; off; off >>= 1) { s += __shfl_xor(s, off); ss += __shfl_xor(ss, off); }
  if (lane == 0) { red[0][wid] = s; red[1][wid] = ss; }
  __syncthreads();
  float ts = 0.f, tss = 0.f;
#pragma unroll
  for (int w = 0; w < 8; ++w) { ts += red[0][w]; tss += red[1][w]; }
  float mu = ts * (1.0f / HID);
  float var = tss * (1.0f / HID) - mu * mu;
  float r = rsqrtf(var + 1e-6f);
  for (int i = tid; i < HID; i += 256) {
    float y = (xr[i] - mu) * r * (1.0f + mod[HID + i]) + mod[i];
    xm[(size_t)row * HID + i] = f2bf(y);
  }
}

// ---------- 4/7) WMMA GEMM: C = A(bf16, MxK) * Bt(bf16, NxK)^T + bias ----------
// Tiles stream global->LDS via async DMA (ASYNCcnt), double-buffered, BK=32.
// MODE 1: epilogue -> qkv f32 (cols<9216) or gelu->bf16 into a2 (cols>=9216)
// MODE 2: epilogue -> out = x + gate * (acc + bias)
template <int MODE>
__global__ __launch_bounds__(256, 1)
void gemm_bf16_kernel(const __bf16* __restrict__ A, const __bf16* __restrict__ Bt,
                      const float* __restrict__ bias, int K,
                      float* __restrict__ qkv, __bf16* __restrict__ a2,
                      const float* __restrict__ x, const float* __restrict__ gate,
                      float* __restrict__ out) {
  __shared__ __bf16 As[2][128 * 40];                // 128 rows x 32 K, stride 40, x2 buffers
  __shared__ __bf16 Bs[2][128 * 40];
  int tid = threadIdx.x;
  int wid = tid >> 5, lane = tid & 31, hl = lane >> 4, ln = lane & 15;
  int wm = wid & 3, wn = wid >> 2;                  // 4 waves on M (32 each), 2 on N (64 each)
  int m0 = blockIdx.y * 128, n0 = blockIdx.x * 128;
  v8f c[2][4] = {};

  // tile copy: thread owns row r, two 16B pieces at half-offsets s2, s2+8
  int r = tid >> 1, s2 = (tid & 1) * 16;
  const __bf16* gA = A  + (size_t)(m0 + r) * K + s2;
  const __bf16* gB = Bt + (size_t)(n0 + r) * K + s2;
  unsigned aAs[2] = { ldsaddr(&As[0][r * 40 + s2]), ldsaddr(&As[1][r * 40 + s2]) };
  unsigned aBs[2] = { ldsaddr(&Bs[0][r * 40 + s2]), ldsaddr(&Bs[1][r * 40 + s2]) };

  auto issue = [&](int k0, int buf) {
    async_cp16(aAs[buf],      gA + k0);
    async_cp16(aAs[buf] + 16, gA + k0 + 8);
    async_cp16(aBs[buf],      gB + k0);
    async_cp16(aBs[buf] + 16, gB + k0 + 8);
  };

  int nstep = K / 32;
  issue(0, 0);
  for (int step = 0; step < nstep; ++step) {
    int cur = step & 1;
    wait_async0();                                  // this wave's tile DMA complete
    __syncthreads();                                // all waves' tile DMA complete
    if (step + 1 < nstep) issue((step + 1) * 32, cur ^ 1);  // stream next tile during compute

    v16bf a[2];
#pragma unroll
    for (int mi = 0; mi < 2; ++mi) {
      const __bf16* ap = &As[cur][(wm * 32 + mi * 16 + ln) * 40];
      a[mi] = ld_ab(ap + 8 * hl, ap + 16 + 8 * hl);
    }
#pragma unroll
    for (int ni = 0; ni < 4; ++ni) {
      const __bf16* bp = &Bs[cur][(wn * 64 + ni * 16 + ln) * 40];
      v16bf b = ld_ab(bp + 8 * hl, bp + 16 + 8 * hl);
      c[0][ni] = wmma_bf16(a[0], b, c[0][ni]);
      c[1][ni] = wmma_bf16(a[1], b, c[1][ni]);
    }
  }

#pragma unroll
  for (int mi = 0; mi < 2; ++mi)
#pragma unroll
    for (int ni = 0; ni < 4; ++ni)
#pragma unroll
      for (int rr = 0; rr < 8; ++rr) {
        int gm = m0 + wm * 32 + mi * 16 + rr + 8 * hl;
        int gn = n0 + wn * 64 + ni * 16 + ln;
        float v = c[mi][ni][rr] + bias[gn];
        if (MODE == 1) {
          if (gn < QKV_N) qkv[(size_t)gm * QKV_N + gn] = v;
          else            a2[(size_t)gm * K2 + HID + (gn - QKV_N)] = f2bf(gelu_tanh(v));
        } else {
          size_t o = (size_t)gm * HID + gn;
          out[o] = x[o] + gate[gn] * v;
        }
      }
}

// ---------- 5) qkv post: rmsnorm + rope + layout for attention ----------
__global__ void qkvpost_kernel(const float* __restrict__ qkv,
                               const float* __restrict__ qw, const float* __restrict__ kw,
                               const float* __restrict__ rc, const float* __restrict__ rs,
                               __bf16* __restrict__ qb, __bf16* __restrict__ kb,
                               __bf16* __restrict__ vt) {
  int h = blockIdx.x, s = blockIdx.y, t = threadIdx.x;   // 32 threads, 4 d each
  const float* base = qkv + (size_t)s * QKV_N + h * HD;
  float4 q = *(const float4*)(base + 4 * t);
  float4 k = *(const float4*)(base + HID + 4 * t);
  float4 v = *(const float4*)(base + 2 * HID + 4 * t);

  float sq = q.x * q.x + q.y * q.y + q.z * q.z + q.w * q.w;
  float sk = k.x * k.x + k.y * k.y + k.z * k.z + k.w * k.w;
  for (int off = 16; off; off >>= 1) { sq += __shfl_xor(sq, off); sk += __shfl_xor(sk, off); }
  float rq = rsqrtf(sq * (1.0f / HD) + 1e-6f);
  float rk = rsqrtf(sk * (1.0f / HD) + 1e-6f);

  float qd[4] = { q.x * rq * qw[4 * t + 0], q.y * rq * qw[4 * t + 1],
                  q.z * rq * qw[4 * t + 2], q.w * rq * qw[4 * t + 3] };
  float kd[4] = { k.x * rk * kw[4 * t + 0], k.y * rk * kw[4 * t + 1],
                  k.z * rk * kw[4 * t + 2], k.w * rk * kw[4 * t + 3] };

  if (s < IMGL) {                                   // rotary on image tokens
    float c0 = rc[(size_t)s * 64 + 2 * t],     s0 = rs[(size_t)s * 64 + 2 * t];
    float c1 = rc[(size_t)s * 64 + 2 * t + 1], s1 = rs[(size_t)s * 64 + 2 * t + 1];
    float a0 = qd[0], a1 = qd[1], a2 = qd[2], a3 = qd[3];
    qd[0] = a0 * c0 - a1 * s0; qd[1] = a0 * s0 + a1 * c0;
    qd[2] = a2 * c1 - a3 * s1; qd[3] = a2 * s1 + a3 * c1;
    float b0 = kd[0], b1 = kd[1], b2 = kd[2], b3 = kd[3];
    kd[0] = b0 * c0 - b1 * s0; kd[1] = b0 * s0 + b1 * c0;
    kd[2] = b2 * c1 - b3 * s1; kd[3] = b2 * s1 + b3 * c1;
  }
  const float qscale = 0.08838834764831845f;        // 1/sqrt(128), folded into q
  size_t qo = ((size_t)h * SEQ + s) * HD + 4 * t;
  float vv[4] = { v.x, v.y, v.z, v.w };
#pragma unroll
  for (int i = 0; i < 4; ++i) {
    qb[qo + i] = f2bf(qd[i] * qscale);
    kb[qo + i] = f2bf(kd[i]);
    vt[((size_t)h * HD + 4 * t + i) * SEQ + s] = f2bf(vv[i]);  // transposed V
  }
}

// ---------- 6) flash attention with WMMA ----------
__global__ __launch_bounds__(256, 1)
void attn_kernel(const __bf16* __restrict__ qb, const __bf16* __restrict__ kb,
                 const __bf16* __restrict__ vt, __bf16* __restrict__ a2) {
  __shared__ __bf16 Ks[64 * 136];                  // 64 keys x 128 d (stride 136)
  __shared__ __bf16 Vs[128 * 72];                  // 128 d x 64 keys (stride 72)
  __shared__ __bf16 Ps[8 * 16 * 72];               // per-wave P tile 16x64 (stride 72)
  int head = blockIdx.y;
  int q0 = blockIdx.x * 128;
  int tid = threadIdx.x;
  int wid = tid >> 5, lane = tid & 31, hl = lane >> 4, ln = lane & 15;
  const __bf16* qh = qb + (size_t)head * SEQ * HD;
  const __bf16* kh = kb + (size_t)head * SEQ * HD;
  const __bf16* vh = vt + (size_t)head * HD * SEQ;

  // Q fragments held in registers (direct global loads, A-layout)
  int qrow = q0 + wid * 16 + ln;
  const __bf16* qp = qh + (size_t)qrow * HD;
  v16bf aq[4];
#pragma unroll
  for (int kc = 0; kc < 4; ++kc)
    aq[kc] = ld_ab(qp + kc * 32 + 8 * hl, qp + kc * 32 + 16 + 8 * hl);

  v8f o[8] = {};
  float m_[8], l_[8];
#pragma unroll
  for (int r = 0; r < 8; ++r) { m_[r] = -1e30f; l_[r] = 0.0f; }

  // tile copy addressing: K tile rows (rk, rk+32); V tile rows (rv, rv+64)
  int rk = tid >> 3, sk_ = tid & 7;
  int rv = tid >> 2, sv_ = tid & 3;
  const __bf16* gK0 = kh + (size_t)rk * HD + sk_ * 16;
  const __bf16* gK1 = gK0 + (size_t)32 * HD;
  const __bf16* gV0 = vh + (size_t)rv * SEQ + sv_ * 16;
  const __bf16* gV1 = gV0 + (size_t)64 * SEQ;
  __bf16* sK0 = &Ks[rk * 136 + sk_ * 16];
  __bf16* sK1 = &Ks[(rk + 32) * 136 + sk_ * 16];
  __bf16* sV0 = &Vs[rv * 72 + sv_ * 16];
  __bf16* sV1 = &Vs[(rv + 64) * 72 + sv_ * 16];

  uint4 rkv[8];
  auto fetch = [&](int t) {
    size_t koff = (size_t)t * 64 * HD;             // K tile advances by 64 rows
    size_t voff = (size_t)t * 64;                  // V tile advances by 64 cols
    const uint4* k0p = (const uint4*)(gK0 + koff);
    const uint4* k1p = (const uint4*)(gK1 + koff);
    const uint4* v0p = (const uint4*)(gV0 + voff);
    const uint4* v1p = (const uint4*)(gV1 + voff);
    rkv[0] = k0p[0]; rkv[1] = k0p[1]; rkv[2] = k1p[0]; rkv[3] = k1p[1];
    rkv[4] = v0p[0]; rkv[5] = v0p[1]; rkv[6] = v1p[0]; rkv[7] = v1p[1];
  };

  fetch(0);
  for (int t = 0; t < SEQ / 64; ++t) {
    ((uint4*)sK0)[0] = rkv[0]; ((uint4*)sK0)[1] = rkv[1];
    ((uint4*)sK1)[0] = rkv[2]; ((uint4*)sK1)[1] = rkv[3];
    ((uint4*)sV0)[0] = rkv[4]; ((uint4*)sV0)[1] = rkv[5];
    ((uint4*)sV1)[0] = rkv[6]; ((uint4*)sV1)[1] = rkv[7];
    __syncthreads();
    if (t + 1 < SEQ / 64) fetch(t + 1);            // overlap next K/V tile with compute

    // S = Q K^T  (4 N-frags of 16 keys, K-loop over d=128)
    v8f s[4] = {};
#pragma unroll
    for (int nf = 0; nf < 4; ++nf) {
      const __bf16* kp = &Ks[(nf * 16 + ln) * 136];
#pragma unroll
      for (int kc = 0; kc < 4; ++kc) {
        const __bf16* bp = kp + kc * 32;
        v16bf b = ld_ab(bp + 8 * hl, bp + 16 + 8 * hl);
        s[nf] = wmma_bf16(aq[kc], b, s[nf]);
      }
    }

    // online softmax per row
#pragma unroll
    for (int r = 0; r < 8; ++r) {
      float mx = fmaxf(fmaxf(s[0][r], s[1][r]), fmaxf(s[2][r], s[3][r]));
      mx = fmaxf(mx, __shfl_xor(mx, 1));
      mx = fmaxf(mx, __shfl_xor(mx, 2));
      mx = fmaxf(mx, __shfl_xor(mx, 4));
      mx = fmaxf(mx, __shfl_xor(mx, 8));
      float mn = fmaxf(m_[r], mx);
      float sc = __expf(m_[r] - mn);
      m_[r] = mn;
      float rsum = 0.0f;
#pragma unroll
      for (int nf = 0; nf < 4; ++nf) {
        float p = __expf(s[nf][r] - mn);
        s[nf][r] = p;
        rsum += p;
      }
      rsum += __shfl_xor(rsum, 1);
      rsum += __shfl_xor(rsum, 2);
      rsum += __shfl_xor(rsum, 4);
      rsum += __shfl_xor(rsum, 8);
      l_[r] = l_[r] * sc + rsum;
#pragma unroll
      for (int nf2 = 0; nf2 < 8; ++nf2) o[nf2][r] *= sc;
    }

    // P through LDS: C-layout -> A-layout (wave-private slice)
#pragma unroll
    for (int nf = 0; nf < 4; ++nf)
#pragma unroll
      for (int r = 0; r < 8; ++r)
        Ps[wid * 16 * 72 + (r + 8 * hl) * 72 + nf * 16 + ln] = f2bf(s[nf][r]);
    __syncthreads();

    const __bf16* pp = &Ps[wid * 16 * 72 + ln * 72];
    v16bf pa[2];
    pa[0] = ld_ab(pp + 8 * hl, pp + 16 + 8 * hl);
    pa[1] = ld_ab(pp + 32 + 8 * hl, pp + 48 + 8 * hl);

#pragma unroll
    for (int nf2 = 0; nf2 < 8; ++nf2) {
      const __bf16* vp = &Vs[(nf2 * 16 + ln) * 72];
#pragma unroll
      for (int kc2 = 0; kc2 < 2; ++kc2) {
        const __bf16* bp = vp + kc2 * 32;
        v16bf bv = ld_ab(bp + 8 * hl, bp + 16 + 8 * hl);
        o[nf2] = wmma_bf16(pa[kc2], bv, o[nf2]);
      }
    }
    __syncthreads();                               // protect Ks/Vs/Ps before next stash
  }

  // write attn output (bf16) into GEMM2 A-buffer columns [head*128, head*128+128)
#pragma unroll
  for (int r = 0; r < 8; ++r) {
    float inv = 1.0f / l_[r];
    int gm = q0 + wid * 16 + r + 8 * hl;
    size_t ro = (size_t)gm * K2 + head * HD;
#pragma unroll
    for (int nf2 = 0; nf2 < 8; ++nf2)
      a2[ro + nf2 * 16 + ln] = f2bf(o[nf2][r] * inv);
  }
}

// ---------- host launcher ----------
extern "C" void kernel_launch(void* const* d_in, const int* in_sizes, int n_in,
                              void* d_out, int out_size, void* d_ws, size_t ws_size,
                              hipStream_t stream) {
  const float* x     = (const float*)d_in[0];
  const float* vec   = (const float*)d_in[1];
  const float* w_mod = (const float*)d_in[2];
  const float* b_mod = (const float*)d_in[3];
  const float* w1    = (const float*)d_in[4];
  const float* b1    = (const float*)d_in[5];
  const float* w2    = (const float*)d_in[6];
  const float* b2    = (const float*)d_in[7];
  const float* qw    = (const float*)d_in[8];
  const float* kw    = (const float*)d_in[9];
  const float* rc    = (const float*)d_in[10];
  const float* rsn   = (const float*)d_in[11];
  float* out = (float*)d_out;

  // workspace carve (~420 MB)
  char* p = (char*)d_ws;
  float*  mod  = (float*)p;  p += (size_t)QKV_N * 4;
  __bf16* xm   = (__bf16*)p; p += (size_t)SEQ * HID * 2;
  __bf16* w1t  = (__bf16*)p; p += (size_t)N1 * HID * 2;
  __bf16* w2t  = (__bf16*)p; p += (size_t)HID * K2 * 2;
  float*  qkv  = (float*)p;  p += (size_t)SEQ * QKV_N * 4;
  __bf16* qb   = (__bf16*)p; p += (size_t)HEADS * SEQ * HD * 2;
  __bf16* kb   = (__bf16*)p; p += (size_t)HEADS * SEQ * HD * 2;
  __bf16* vt   = (__bf16*)p; p += (size_t)HEADS * HD * SEQ * 2;
  __bf16* a2   = (__bf16*)p; p += (size_t)SEQ * K2 * 2;

  // 1) modulation vector
  zero_mod_kernel<<<QKV_N / 256, 256, 0, stream>>>(mod);
  mod_gemv_kernel<<<dim3(QKV_N / 128, 16), 128, 0, stream>>>(vec, w_mod, b_mod, mod);

  // 2) weight transpose+convert
  transcvt_kernel<<<dim3(N1 / 32, HID / 32), dim3(32, 8), 0, stream>>>(w1, w1t, HID, N1);
  transcvt_kernel<<<dim3(HID / 32, K2 / 32), dim3(32, 8), 0, stream>>>(w2, w2t, K2, HID);

  // 3) layernorm + modulation
  ln_mod_kernel<<<SEQ, 256, 0, stream>>>(x, mod, xm);

  // 4) GEMM1: x_mod @ w1 (+b1) -> qkv f32, gelu(mlp) bf16
  gemm_bf16_kernel<1><<<dim3(N1 / 128, SEQ / 128), 256, 0, stream>>>(
      xm, w1t, b1, HID, qkv, a2, nullptr, nullptr, nullptr);

  // 5) rmsnorm + rope + attention layouts
  qkvpost_kernel<<<dim3(HEADS, SEQ), 32, 0, stream>>>(qkv, qw, kw, rc, rsn, qb, kb, vt);

  // 6) attention
  attn_kernel<<<dim3(SEQ / 128, HEADS), 256, 0, stream>>>(qb, kb, vt, a2);

  // 7) GEMM2: [attn | gelu(mlp)] @ w2 (+b2), residual + gate
  gemm_bf16_kernel<2><<<dim3(HID / 128, SEQ / 128), 256, 0, stream>>>(
      a2, w2t, b2, K2, nullptr, nullptr, x, mod + 2 * HID, out);

  (void)in_sizes; (void)n_in; (void)out_size; (void)ws_size;
}
